// ScanLayer_53369263620671
// MI455X (gfx1250) — compile-verified, matrix operand order checked by reading, not run on hardware
//
#include <hip/hip_runtime.h>
#include <math.h>

// Problem constants (from reference setup_inputs)
#define BS   4
#define SEQ  2048
#define CC   8
#define DD   256
#define EE   512
#define E2   1024   // 2E

typedef __attribute__((ext_vector_type(16))) _Float16 v16h;
typedef __attribute__((ext_vector_type(8)))  float    v8f;

// CDNA5 16-bit A-matrix 16x32 fragment position:
// lane = m + 16*half,  element j = 2*v + parity,
// k = base(v) + 8*half + parity, base(v) = v<4 ? 2v : 16+2(v-4)
__device__ __forceinline__ void frag_pos(int kk, int maj, int& lane, int& j) {
  int half = (kk >> 3) & 1;
  int v    = ((kk >> 4) << 2) + ((kk & 7) >> 1);
  lane = maj + (half << 4);
  j    = (v << 1) + (kk & 1);
}

__device__ __forceinline__ float fast_sigmoid(float x) {
  return __builtin_amdgcn_rcpf(1.0f + __expf(-x));
}

// ---- Weight repack: f32 [C,D,2E] -> f16 B-fragments [c][kt][nt][lane][16] ----
__global__ void __launch_bounds__(256) prep_win(const float* __restrict__ W,
                                                _Float16* __restrict__ Wf) {
  int t    = blockIdx.x * 256 + threadIdx.x;      // < 2^21
  int j    = t & 15;
  int lane = (t >> 4) & 31;
  int nt   = (t >> 9) & 63;                       // 2E/16 = 64
  int kt   = (t >> 15) & 7;                       // D/32  = 8
  int c    = t >> 18;
  int n    = nt * 16 + (lane & 15);
  int half = lane >> 4;
  int v    = j >> 1, par = j & 1;
  int base = (v < 4) ? (2 * v) : (16 + 2 * (v - 4));
  int k    = kt * 32 + base + 8 * half + par;
  Wf[t] = (_Float16)W[((size_t)c * DD + k) * E2 + n];
}

// ---- Weight repack: f32 [C,E,D] -> f16 B-fragments ----
__global__ void __launch_bounds__(256) prep_wout(const float* __restrict__ W,
                                                 _Float16* __restrict__ Wf) {
  int t    = blockIdx.x * 256 + threadIdx.x;      // < 2^20
  int j    = t & 15;
  int lane = (t >> 4) & 31;
  int nt   = (t >> 9) & 15;                       // D/16 = 16
  int kt   = (t >> 13) & 15;                      // E/32 = 16
  int c    = t >> 17;
  int n    = nt * 16 + (lane & 15);
  int half = lane >> 4;
  int v    = j >> 1, par = j & 1;
  int base = (v < 4) ? (2 * v) : (16 + 2 * (v - 4));
  int k    = kt * 32 + base + 8 * half + par;
  Wf[t] = (_Float16)W[((size_t)c * EE + k) * DD + n];
}

// ---- Kernel 1: LayerNorm + proj_in GEMM (WMMA) + sigmoid/silu gates ----
// M-tile = 64 rows. grid: 1024 blocks (128 row-groups x 8 columns), 256 threads.
__global__ void __launch_bounds__(256) k_proj_in(
    const float* __restrict__ x, const float* __restrict__ gamma,
    const float* __restrict__ beta, const float* __restrict__ b_in,
    const _Float16* __restrict__ Wi, float* __restrict__ aBuf,
    float* __restrict__ bBuf) {
  __shared__ v16h afragv[32 * 32];                // 4 rowgroups x 8 kt : 32 KB
  _Float16* afrag = (_Float16*)afragv;
  __shared__ float sSum[64 * 4], sSq[64 * 4];
  __shared__ float sMean[64], sRstd[64];

  int c  = blockIdx.x & 7;
  int rg = blockIdx.x >> 3;
  int r0 = rg * 64;
  int b  = r0 / SEQ;
  int n0 = r0 % SEQ;
  int t  = threadIdx.x;
  int m  = t >> 2;                                 // row 0..63
  int q  = t & 3;                                  // quarter of row (64 floats)

  const float* xrow = x + ((size_t)(b * SEQ + n0 + m) * CC + c) * DD + q * 64;
  const float4* x4  = (const float4*)xrow;

  // Pass 1: partial sums for LayerNorm stats
  {
    float s = 0.f, ss = 0.f;
    #pragma unroll
    for (int i = 0; i < 16; ++i) {
      float4 v = x4[i];
      s  += v.x + v.y + v.z + v.w;
      ss += v.x * v.x + v.y * v.y + v.z * v.z + v.w * v.w;
    }
    sSum[m * 4 + q] = s;
    sSq[m * 4 + q]  = ss;
  }
  __syncthreads();
  if (t < 64) {
    float s  = sSum[t * 4] + sSum[t * 4 + 1] + sSum[t * 4 + 2] + sSum[t * 4 + 3];
    float ss = sSq[t * 4] + sSq[t * 4 + 1] + sSq[t * 4 + 2] + sSq[t * 4 + 3];
    float mean = s * (1.0f / DD);
    float var  = ss * (1.0f / DD) - mean * mean;
    sMean[t] = mean;
    sRstd[t] = rsqrtf(var + 1e-5f);
  }
  __syncthreads();

  // Pass 2: normalize + affine -> f16 A-fragments in LDS
  {
    float mean = sMean[m], rstd = sRstd[m];
    int mm = m & 15, rgrp = m >> 4;
    for (int i4 = 0; i4 < 16; ++i4) {
      float4 v = x4[i4];
      float vv[4] = {v.x, v.y, v.z, v.w};
      #pragma unroll
      for (int e = 0; e < 4; ++e) {
        int d = q * 64 + i4 * 4 + e;
        float val = (vv[e] - mean) * rstd * gamma[c * DD + d] + beta[c * DD + d];
        int kt = d >> 5, kk = d & 31;
        int lane, j; frag_pos(kk, mm, lane, j);
        afrag[((rgrp * 8 + kt) * 32 + lane) * 16 + j] = (_Float16)val;
      }
    }
  }
  __syncthreads();

  // GEMM: 8 waves x 8 nt-tiles each; B fragments double-buffered and
  // reused across 4 rowgroup accumulators.
  int wave = t >> 5, lane = t & 31, ncol = lane & 15, hh = lane >> 4;
  const v16h* Wiv = (const v16h*)Wi;
  auto ldB = [&](int nt, int kc, v16h bf[4]) {
    #pragma unroll
    for (int k = 0; k < 4; ++k)
      bf[k] = Wiv[((size_t)(c * 8 + kc * 4 + k) * 64 + nt) * 32 + lane];
  };

  v16h buf0[4], buf1[4];
  ldB(wave * 8, 0, buf0);                          // prologue
  for (int sidx = 0; sidx < 8; ++sidx) {
    int nt = wave * 8 + sidx;                      // 0..63
    float biasv = b_in[c * E2 + nt * 16 + ncol];
    v8f acc[4];
    #pragma unroll
    for (int r = 0; r < 4; ++r)
      #pragma unroll
      for (int g = 0; g < 8; ++g) acc[r][g] = biasv;

    ldB(nt, 1, buf1);                              // overlap with chunk 0 math
    #pragma unroll
    for (int r = 0; r < 4; ++r)
      #pragma unroll
      for (int k = 0; k < 4; ++k)
        acc[r] = __builtin_amdgcn_wmma_f32_16x16x32_f16(
            false, afragv[(r * 8 + k) * 32 + lane], false, buf0[k],
            (short)0, acc[r], false, false);

    if (sidx < 7) ldB(nt + 1, 0, buf0);            // prefetch next tile
    #pragma unroll
    for (int r = 0; r < 4; ++r)
      #pragma unroll
      for (int k = 0; k < 4; ++k)
        acc[r] = __builtin_amdgcn_wmma_f32_16x16x32_f16(
            false, afragv[(r * 8 + 4 + k) * 32 + lane], false, buf1[k],
            (short)0, acc[r], false, false);

    // Wave-uniform gate selection (nt < 32 <=> all 16 columns < E)
    if (__builtin_amdgcn_readfirstlane(nt) < 32) {
      int e = nt * 16 + ncol;
      #pragma unroll
      for (int r = 0; r < 4; ++r)
        #pragma unroll
        for (int g = 0; g < 8; ++g) {
          int mrow = r * 16 + g + 8 * hh;
          aBuf[((size_t)(b * SEQ + n0 + mrow) * CC + c) * EE + e] =
              fast_sigmoid(acc[r][g]);
        }
    } else {
      int e = nt * 16 + ncol - EE;
      #pragma unroll
      for (int r = 0; r < 4; ++r)
        #pragma unroll
        for (int g = 0; g < 8; ++g) {
          int mrow = r * 16 + g + 8 * hh;
          float xg = acc[r][g];
          bBuf[((size_t)(b * SEQ + n0 + mrow) * CC + c) * EE + e] =
              xg * fast_sigmoid(xg);
        }
    }
  }
}

// ---- Kernel 2: linear recurrence h_t = a_t*h_{t-1} + b_t (h0 folded) ----
__global__ void __launch_bounds__(256) k_scan(const float* __restrict__ h_prev,
                                              float* __restrict__ aBuf,
                                              const float* __restrict__ bBuf,
                                              float* __restrict__ hLast) {
  int t = blockIdx.x * 256 + threadIdx.x;          // < 16384
  int e = t & (EE - 1);
  int c = (t >> 9) & (CC - 1);
  int b = t >> 12;
  float h = h_prev[((size_t)b * CC + c) * EE + e];
  size_t base = (((size_t)b * SEQ) * CC + c) * EE + e;
  const size_t stride = (size_t)CC * EE;
  #pragma unroll 4
  for (int n = 0; n < SEQ; ++n) {
    size_t idx = base + (size_t)n * stride;
    float a  = aBuf[idx];
    float bb = bBuf[idx];
    h = fmaf(a, h, bb);
    aBuf[idx] = h;                                 // h replaces a in place
  }
  hLast[((size_t)b * CC + c) * EE + e] = h;
}

// ---- Kernel 3: proj_out GEMM (WMMA) + bias + residual ----
// M-tile = 64 rows. grid: 1024 blocks, 256 threads.
__global__ void __launch_bounds__(256) k_proj_out(
    const float* __restrict__ hBuf, const float* __restrict__ x,
    const float* __restrict__ b_out, const _Float16* __restrict__ Wo,
    float* __restrict__ out) {
  __shared__ v16h hfragv[64 * 32];                 // 4 rowgroups x 16 kt : 64 KB
  _Float16* hfrag = (_Float16*)hfragv;

  int c  = blockIdx.x & 7;
  int rg = blockIdx.x >> 3;
  int r0 = rg * 64;
  int b  = r0 / SEQ;
  int n0 = r0 % SEQ;
  int t  = threadIdx.x;

  // h (f32) -> f16 A-fragments: thread handles row m, quarter q (128 floats)
  {
    int m = t >> 2, q = t & 3;
    int mm = m & 15, rgrp = m >> 4;
    const float4* h4 =
        (const float4*)(hBuf + ((size_t)(b * SEQ + n0 + m) * CC + c) * EE + q * 128);
    for (int i4 = 0; i4 < 32; ++i4) {
      float4 v = h4[i4];
      float vv[4] = {v.x, v.y, v.z, v.w};
      #pragma unroll
      for (int e = 0; e < 4; ++e) {
        int k = q * 128 + i4 * 4 + e;
        int kt = k >> 5, kk = k & 31;
        int lane, j; frag_pos(kk, mm, lane, j);
        hfrag[((rgrp * 16 + kt) * 32 + lane) * 16 + j] = (_Float16)vv[e];
      }
    }
  }
  __syncthreads();

  int wave = t >> 5, lane = t & 31, ncol = lane & 15, hh = lane >> 4;
  const v16h* Wov = (const v16h*)Wo;
  auto ldB = [&](int nt, int kc, v16h bf[4]) {
    #pragma unroll
    for (int k = 0; k < 4; ++k)
      bf[k] = Wov[((size_t)(c * 16 + kc * 4 + k) * 16 + nt) * 32 + lane];
  };
  auto mm16 = [&](int kc, const v16h bf[4], v8f acc[4]) {
    #pragma unroll
    for (int r = 0; r < 4; ++r)
      #pragma unroll
      for (int k = 0; k < 4; ++k)
        acc[r] = __builtin_amdgcn_wmma_f32_16x16x32_f16(
            false, hfragv[(r * 16 + kc * 4 + k) * 32 + lane], false, bf[k],
            (short)0, acc[r], false, false);
  };

  v16h buf0[4], buf1[4];
  ldB(wave, 0, buf0);                              // prologue
  for (int sidx = 0; sidx < 2; ++sidx) {
    int nt = wave + 8 * sidx;                      // 0..15
    float biasv = b_out[c * DD + nt * 16 + ncol];
    v8f acc[4];
    #pragma unroll
    for (int r = 0; r < 4; ++r)
      #pragma unroll
      for (int g = 0; g < 8; ++g) acc[r][g] = biasv;

    ldB(nt, 1, buf1);
    mm16(0, buf0, acc);
    ldB(nt, 2, buf0);
    mm16(1, buf1, acc);
    ldB(nt, 3, buf1);
    mm16(2, buf0, acc);
    if (sidx == 0) ldB(wave + 8, 0, buf0);         // prefetch next tile
    mm16(3, buf1, acc);

    int d = nt * 16 + ncol;
    #pragma unroll
    for (int r = 0; r < 4; ++r)
      #pragma unroll
      for (int g = 0; g < 8; ++g) {
        int mrow = r * 16 + g + 8 * hh;
        size_t oi = ((size_t)(b * SEQ + n0 + mrow) * CC + c) * DD + d;
        out[oi] = acc[r][g] + x[oi];               // residual
      }
  }
}

extern "C" void kernel_launch(void* const* d_in, const int* in_sizes, int n_in,
                              void* d_out, int out_size, void* d_ws, size_t ws_size,
                              hipStream_t stream) {
  (void)in_sizes; (void)n_in; (void)out_size; (void)ws_size;
  const float* x      = (const float*)d_in[0];
  const float* h_prev = (const float*)d_in[1];
  const float* gamma  = (const float*)d_in[2];
  const float* beta   = (const float*)d_in[3];
  const float* W_in   = (const float*)d_in[4];
  const float* b_in   = (const float*)d_in[5];
  const float* W_out  = (const float*)d_in[6];
  const float* b_out  = (const float*)d_in[7];
  float* out = (float*)d_out;

  const size_t actElems = (size_t)BS * SEQ * CC * EE;   // 33.5M
  float*    aBuf = (float*)d_ws;
  float*    bBuf = aBuf + actElems;
  _Float16* Wi16 = (_Float16*)(bBuf + actElems);
  _Float16* Wo16 = Wi16 + (size_t)CC * DD * E2;

  // Repack weights to f16 fragment layout (deterministic every call)
  prep_win <<<(CC * DD * E2) / 256, 256, 0, stream>>>(W_in,  Wi16);
  prep_wout<<<(CC * EE * DD) / 256, 256, 0, stream>>>(W_out, Wo16);

  // LN + proj_in + gates (64-row M-tiles)
  k_proj_in<<<(BS * SEQ / 64) * CC, 256, 0, stream>>>(x, gamma, beta, b_in,
                                                      Wi16, aBuf, bBuf);
  // Recurrence; h_last goes after the main output block
  float* hLast = out + (size_t)BS * SEQ * CC * DD;
  k_scan<<<(BS * CC * EE) / 256, 256, 0, stream>>>(h_prev, aBuf, bBuf, hLast);

  // proj_out + residual (64-row M-tiles)
  k_proj_out<<<(BS * SEQ / 64) * CC, 256, 0, stream>>>(aBuf, x, b_out, Wo16, out);
}
